// TCPHeaderTransformer_48825188221432
// MI455X (gfx1250) — matrix operand order repaired and checked by reading, not compiled, not run on hardware
//
#include <hip/hip_runtime.h>
#include <hip/hip_bf16.h>
#include <math.h>

// ---------------------------------------------------------------------------
// MI455X (gfx1250) linear-attention transformer forward.
// GEMMs: v_wmma_f32_16x16x32_bf16, 128x64 block tile, 8 waves, 4 wmma per
// wave per K-step. A staged row-major (b128 frag loads), B staged pair-major
// so each WMMA source VGPR is one ds_load_b32. Final 4096x65546 projection is
// HBM-bound (1.07 GB fp32 logits ~= 46us @ 23.3 TB/s) -> bf16 matrix cores
// keep compute under the memory floor. Attention cumsum: chunked scan
// (critical path 1024 -> 64+16) in fp32 VALU (~67 MFLOP total, negligible).
// ---------------------------------------------------------------------------

typedef __attribute__((ext_vector_type(16))) __bf16          v16bf;
typedef __attribute__((ext_vector_type(8)))  unsigned int    v8ui;
typedef __attribute__((ext_vector_type(8)))  float           v8f;

#define D_MODEL   256
#define NHEAD     8
#define HEAD_DIM  32
#define L_SEQ     1024
#define BATCH     4
#define ROWS      (BATCH * L_SEQ)   // 4096
#define D_FF      1024
#define VOCAB     65546
#define PAD_IDX   65545
#define NLAYERS   8

__device__ __forceinline__ float wredsum(float v) {
  #pragma unroll
  for (int o = 16; o > 0; o >>= 1) v += __shfl_xor(v, o, 32);
  return v;
}

__device__ __forceinline__ unsigned short f2bf(float f) {
  unsigned int u = __float_as_uint(f);
  u += 0x7fffu + ((u >> 16) & 1u);           // round-to-nearest-even
  return (unsigned short)(u >> 16);
}

__device__ __forceinline__ float elu1(float x) {
  return x > 0.f ? x + 1.f : __expf(x);      // elu(x)+1
}

// ---------------------------------------------------------------- weights ---
__global__ void f32_to_bf16_kernel(const float* __restrict__ in,
                                   unsigned short* __restrict__ out, int n) {
  int i = blockIdx.x * blockDim.x + threadIdx.x;
  int stride = gridDim.x * blockDim.x;
  for (; i < n; i += stride) out[i] = f2bf(in[i]);
}

// ------------------------------------------------------------- embedding ---
__global__ void embed_kernel(const int* __restrict__ x,
                             const float* __restrict__ emb,
                             float* __restrict__ h, float* __restrict__ mask) {
  int row = blockIdx.x, c = threadIdx.x;
  int tok = x[row];
  h[(size_t)row * D_MODEL + c] = emb[(size_t)tok * D_MODEL + c];
  if (c == 0) mask[row] = (tok == PAD_IDX) ? 0.f : 1.f;
}

// ------------------------------------------------------------- layernorm ---
__global__ __launch_bounds__(256)
void ln_bf16_kernel(const float* __restrict__ x, const float* __restrict__ g,
                    const float* __restrict__ b, unsigned short* __restrict__ out) {
  int lane = threadIdx.x & 31, wave = threadIdx.x >> 5;
  int row = blockIdx.x * 8 + wave;
  const float* xr = x + (size_t)row * D_MODEL;
  float v[8];
  float s = 0.f;
  #pragma unroll
  for (int i = 0; i < 8; ++i) { v[i] = xr[i * 32 + lane]; s += v[i]; }
  s = wredsum(s);
  float mu = s * (1.f / 256.f);
  float qs = 0.f;
  #pragma unroll
  for (int i = 0; i < 8; ++i) { float d = v[i] - mu; qs += d * d; }
  qs = wredsum(qs);
  float rs = rsqrtf(qs * (1.f / 256.f) + 1e-5f);
  unsigned short* orow = out + (size_t)row * D_MODEL;
  #pragma unroll
  for (int i = 0; i < 8; ++i) {
    int c = i * 32 + lane;
    orow[c] = f2bf((v[i] - mu) * rs * g[c] + b[c]);
  }
}

// ------------------------------------------------------------------ RoPE ---
__global__ __launch_bounds__(256)
void rope_kernel(float* __restrict__ q, float* __restrict__ k) {
  int tid = threadIdx.x;
  int row = blockIdx.x * 2 + (tid >> 7);
  int p = tid & 127;
  int hd = p >> 4, i = p & 15;
  int l = row & (L_SEQ - 1);
  float theta = __powf(10000.f, -(float)i / 16.f);
  float ang = (float)l * theta;
  float sn, cs;
  __sincosf(ang, &sn, &cs);
  size_t idx = (size_t)row * D_MODEL + hd * HEAD_DIM + 2 * i;
  float q1 = q[idx], q2 = q[idx + 1];
  q[idx] = q1 * cs - q2 * sn;  q[idx + 1] = q1 * sn + q2 * cs;
  float k1 = k[idx], k2 = k[idx + 1];
  k[idx] = k1 * cs - k2 * sn;  k[idx + 1] = k1 * sn + k2 * cs;
}

// -------------------------------------------------------- bf16 WMMA GEMM ---
// C[M,N] = act(A_bf16[M,K] @ B_bf16[K,N] + bias) [+residual], f32/bf16 out.
// Block: 256 threads / 8 waves, 128x64 tile, K-step 32.
// Wave w: rows (w&3)*32, cols (w>>2)*32 -> 2x2 wmma tiles per K-step.
#define BM 128
#define BN 64
#define BK 32
#define LDA_P 48        // ushorts/row: 96B pitch (16B-aligned segments)
#define LDB_PITCH 136   // ushorts/pair-row: 272B pitch (16B-aligned)

template <bool FULL>
__device__ __forceinline__ void gemm_loop(
    const unsigned short* __restrict__ A, const unsigned short* __restrict__ B,
    const float* __restrict__ bias, const float* __restrict__ residual,
    void* __restrict__ C, int N, int K, int act_gelu, int out_bf16,
    unsigned short* ldsA, unsigned short* ldsB) {
  int tid = threadIdx.x, lane = tid & 31, wave = tid >> 5;
  int mr = wave & 3;        // 32-row group
  int nc = wave >> 2;       // 32-col group
  int bm0 = blockIdx.y * BM;
  int bn0 = blockIdx.x * BN;

  v8f acc[2][2] = {{{}, {}}, {{}, {}}};

  int arow = tid >> 2, aseg = (tid & 3) * 8;   // A: 128 rows x 4 segs (x2)
  int bp = tid >> 4, bn4 = (tid & 15) * 4;     // B: 16 pair-rows x 16 groups

  for (int kk = 0; kk < K; kk += BK) {
    // ---- stage A (128x32): two b128 global loads -> two b128 LDS stores
    {
      const unsigned short* ap = A + (size_t)(bm0 + arow) * K + kk + aseg;
      uint4 av0 = *(const uint4*)ap;
      uint4 av1 = *(const uint4*)(ap + (size_t)64 * K);
      *(uint4*)&ldsA[arow * LDA_P + aseg] = av0;
      *(uint4*)&ldsA[(arow + 64) * LDA_P + aseg] = av1;
    }
    // ---- stage B (32x64) in pair-major: dword = (B[k][n], B[k+1][n])
    {
      unsigned int p0, p1, p2, p3;
      if (FULL) {
        const unsigned short* b0 = B + (size_t)(kk + 2 * bp) * N + bn0 + bn4;
        const unsigned short* b1 = b0 + N;
        unsigned int d0 = *(const unsigned int*)(b0);
        unsigned int d1 = *(const unsigned int*)(b0 + 2);
        unsigned int e0 = *(const unsigned int*)(b1);
        unsigned int e1 = *(const unsigned int*)(b1 + 2);
        p0 = (d0 & 0x0000ffffu) | (e0 << 16);
        p1 = (d0 >> 16)         | (e0 & 0xffff0000u);
        p2 = (d1 & 0x0000ffffu) | (e1 << 16);
        p3 = (d1 >> 16)         | (e1 & 0xffff0000u);
      } else {
        unsigned int pr[4];
        #pragma unroll
        for (int j = 0; j < 4; ++j) {
          int col = bn0 + bn4 + j;
          unsigned int lo = 0, hi = 0;
          if (col < N) {
            lo = B[(size_t)(kk + 2 * bp) * N + col];
            hi = B[(size_t)(kk + 2 * bp + 1) * N + col];
          }
          pr[j] = lo | (hi << 16);
        }
        p0 = pr[0]; p1 = pr[1]; p2 = pr[2]; p3 = pr[3];
      }
      uint4 pv = {p0, p1, p2, p3};
      *(uint4*)&ldsB[bp * LDB_PITCH + bn4 * 2] = pv;
    }
    __syncthreads();

    if (kk + BK < K) {     // global_prefetch_b8 for next tiles
      __builtin_prefetch(A + (size_t)(bm0 + arow) * K + kk + BK + aseg, 0, 1);
      __builtin_prefetch(B + (size_t)(kk + BK + 2 * bp) * N + bn0 + bn4, 0, 1);
    }

    // ---- A fragments: ISA 7.12.2 layout -> two contiguous b128 per frag
    {
      int m = lane & 15, half = lane >> 4;
      v16bf afrag[2];
      #pragma unroll
      for (int mt = 0; mt < 2; ++mt) {
        const unsigned short* ar = &ldsA[(mr * 32 + mt * 16 + m) * LDA_P + half * 8];
        uint4 l128 = *(const uint4*)ar;
        uint4 h128 = *(const uint4*)(ar + 16);
        v8ui au = {l128.x, l128.y, l128.z, l128.w, h128.x, h128.y, h128.z, h128.w};
        afrag[mt] = __builtin_bit_cast(v16bf, au);
      }
      // ---- B fragments: pair-major -> one ds_load_b32 per source VGPR
      int nn = lane & 15, kh = lane >> 4;
      v16bf bfrag[2];
      #pragma unroll
      for (int t = 0; t < 2; ++t) {
        int n = nc * 32 + t * 16 + nn;
        v8ui bu;
        #pragma unroll
        for (int j = 0; j < 8; ++j)
          bu[j] = *(const unsigned int*)&ldsB[(kh * 4 + j + (j & 4)) * LDB_PITCH + 2 * n];
        bfrag[t] = __builtin_bit_cast(v16bf, bu);
      }
      #pragma unroll
      for (int mt = 0; mt < 2; ++mt)
        #pragma unroll
        for (int t = 0; t < 2; ++t)
          acc[mt][t] = __builtin_amdgcn_wmma_f32_16x16x32_bf16(
              false, afrag[mt], false, bfrag[t], (short)0, acc[mt][t], false, false);
    }
    __syncthreads();
  }

  // ---- epilogue: D layout M = r + 8*(lane>=16), N = lane&15
  int half = lane >> 4, nn = lane & 15;
  #pragma unroll
  for (int mt = 0; mt < 2; ++mt) {
    #pragma unroll
    for (int t = 0; t < 2; ++t) {
      int col = bn0 + nc * 32 + t * 16 + nn;
      if (col < N) {
        float bv = bias[col];
        #pragma unroll
        for (int r = 0; r < 8; ++r) {
          int row = bm0 + mr * 32 + mt * 16 + r + 8 * half;
          float vlu = acc[mt][t][r] + bv;
          if (act_gelu) vlu = 0.5f * vlu * (1.f + erff(vlu * 0.70710678118f));
          if (residual) vlu += residual[(size_t)row * N + col];
          if (out_bf16)
            ((unsigned short*)C)[(size_t)row * N + col] = f2bf(vlu);
          else
            ((float*)C)[(size_t)row * N + col] = vlu;
        }
      }
    }
  }
}

__global__ __launch_bounds__(256)
void gemm_bf16_kernel(const unsigned short* __restrict__ A,
                      const unsigned short* __restrict__ B,
                      const float* __restrict__ bias,
                      const float* __restrict__ residual,
                      void* __restrict__ C,
                      int N, int K, int act_gelu, int out_bf16) {
  __shared__ __align__(16) unsigned short ldsA[BM * LDA_P];
  __shared__ __align__(16) unsigned short ldsB[(BK / 2) * LDB_PITCH];
  // Uniform per block; hoists the ragged path entirely out of the K loop.
  if ((int)(blockIdx.x * BN + BN) <= N)
    gemm_loop<true>(A, B, bias, residual, C, N, K, act_gelu, out_bf16, ldsA, ldsB);
  else
    gemm_loop<false>(A, B, bias, residual, C, N, K, act_gelu, out_bf16, ldsA, ldsB);
}

// ---------------------------------------------- chunked linear attention ---
// Thread (d = tid&31, e = tid>>5): d-reductions are wave32 shuffles.

__global__ __launch_bounds__(1024)
void attn_chunk_kernel(const float* __restrict__ k, const float* __restrict__ v,
                       const float* __restrict__ mask,
                       float* __restrict__ kvc, float* __restrict__ zc) {
  int chunk = blockIdx.x, bh = blockIdx.y;
  int b = bh >> 3, hd = bh & 7;
  int tid = threadIdx.x, d = tid & 31, e = tid >> 5;
  __shared__ float pk[32], vv[32];
  float acc = 0.f, zacc = 0.f;
  int rowbase = b * L_SEQ + chunk * 64;
  for (int s2 = 0; s2 < 64; ++s2) {
    int row = rowbase + s2;
    size_t base = (size_t)row * D_MODEL + hd * HEAD_DIM;
    if (tid < 32)        pk[tid]      = elu1(k[base + tid]) * mask[row];
    else if (tid < 64)   vv[tid - 32] = v[base + (tid - 32)];
    __syncthreads();
    acc += pk[d] * vv[e];
    if (e == 0) zacc += pk[d];
    __syncthreads();
  }
  kvc[((size_t)bh * 16 + chunk) * 1024 + d * 32 + e] = acc;
  if (e == 0) zc[((size_t)bh * 16 + chunk) * 32 + d] = zacc;
}

__global__ __launch_bounds__(1024)
void attn_prefix_kernel(float* __restrict__ kvc, float* __restrict__ zc) {
  int bh = blockIdx.x, tid = threadIdx.x;
  float run = 0.f;
  for (int c = 0; c < 16; ++c) {
    size_t idx = ((size_t)bh * 16 + c) * 1024 + tid;
    float t = kvc[idx]; kvc[idx] = run; run += t;
  }
  if (tid < 32) {
    float zr = 0.f;
    for (int c = 0; c < 16; ++c) {
      size_t idx = ((size_t)bh * 16 + c) * 32 + tid;
      float t = zc[idx]; zc[idx] = zr; zr += t;
    }
  }
}

__global__ __launch_bounds__(1024)
void attn_scan_kernel(const float* __restrict__ q, const float* __restrict__ k,
                      const float* __restrict__ v, const float* __restrict__ mask,
                      const float* __restrict__ kvc, const float* __restrict__ zc,
                      float* __restrict__ h) {
  int chunk = blockIdx.x, bh = blockIdx.y;
  int b = bh >> 3, hd = bh & 7;
  int tid = threadIdx.x, d = tid & 31, e = tid >> 5;
  float S  = kvc[((size_t)bh * 16 + chunk) * 1024 + d * 32 + e];
  float Zr = zc[((size_t)bh * 16 + chunk) * 32 + d];
  __shared__ float pq[32], pk[32], vv[32];
  int rowbase = b * L_SEQ + chunk * 64;
  for (int s2 = 0; s2 < 64; ++s2) {
    int row = rowbase + s2;
    size_t base = (size_t)row * D_MODEL + hd * HEAD_DIM;
    if (tid < 32)        pq[tid]      = elu1(q[base + tid]);
    else if (tid < 64)   pk[tid - 32] = elu1(k[base + tid - 32]) * mask[row];
    else if (tid < 96)   vv[tid - 64] = v[base + tid - 64];
    __syncthreads();
    float fk = pk[d];
    S  += fk * vv[e];
    Zr += fk;
    float num = wredsum(pq[d] * S);
    float den = wredsum(pq[d] * Zr) + 1e-6f;
    if (d == 0) h[base + e] += num / den;
    __syncthreads();
  }
}

// ------------------------------------------------------------------ host ---
extern "C" void kernel_launch(void* const* d_in, const int* in_sizes, int n_in,
                              void* d_out, int out_size, void* d_ws, size_t ws_size,
                              hipStream_t stream) {
  (void)in_sizes; (void)n_in; (void)out_size; (void)ws_size;
  // setup_inputs() insertion order:
  //   0: x, 1: emb, per layer 14 leaves:
  //   norm1_g, norm1_b, Wq, bq, Wk, bk, Wv, bv, norm2_g, norm2_b, W1, b1, W2, b2
  //   then out_norm_g, out_norm_b, Wout, bout
  const int*   x   = (const int*)d_in[0];
  const float* emb = (const float*)d_in[1];
  const int LBASE = 2, LSTRIDE = 14;
  const float* out_norm_g = (const float*)d_in[LBASE + NLAYERS * LSTRIDE + 0];
  const float* out_norm_b = (const float*)d_in[LBASE + NLAYERS * LSTRIDE + 1];
  const float* Wout       = (const float*)d_in[LBASE + NLAYERS * LSTRIDE + 2];
  const float* bout       = (const float*)d_in[LBASE + NLAYERS * LSTRIDE + 3];

  char* ws = (char*)d_ws;
  size_t off = 0;
  auto alloc = [&](size_t bytes) -> char* {
    char* p = ws + off;
    off = (off + bytes + 255) & ~(size_t)255;
    return p;
  };
  unsigned short* wqb[NLAYERS]; unsigned short* wkb[NLAYERS];
  unsigned short* wvb[NLAYERS]; unsigned short* w1b[NLAYERS];
  unsigned short* w2b[NLAYERS];
  for (int i = 0; i < NLAYERS; ++i) {
    wqb[i] = (unsigned short*)alloc((size_t)D_MODEL * D_MODEL * 2);
    wkb[i] = (unsigned short*)alloc((size_t)D_MODEL * D_MODEL * 2);
    wvb[i] = (unsigned short*)alloc((size_t)D_MODEL * D_MODEL * 2);
    w1b[i] = (unsigned short*)alloc((size_t)D_MODEL * D_FF * 2);
    w2b[i] = (unsigned short*)alloc((size_t)D_FF * D_MODEL * 2);
  }
  unsigned short* woutb = (unsigned short*)alloc((size_t)D_MODEL * VOCAB * 2);
  float* h            = (float*)alloc((size_t)ROWS * D_MODEL * 4);
  unsigned short* hln = (unsigned short*)alloc((size_t)ROWS * D_MODEL * 2);
  float* qb = (float*)alloc((size_t)ROWS * D_MODEL * 4);
  float* kb = (float*)alloc((size_t)ROWS * D_MODEL * 4);
  float* vb = (float*)alloc((size_t)ROWS * D_MODEL * 4);
  unsigned short* ff1 = (unsigned short*)alloc((size_t)ROWS * D_FF * 2);
  float* kvc  = (float*)alloc((size_t)32 * 16 * 1024 * 4);
  float* zc   = (float*)alloc((size_t)32 * 16 * 32 * 4);
  float* mask = (float*)alloc((size_t)ROWS * 4);

  auto conv = [&](const float* src, unsigned short* dst, int n) {
    int blocks = (n + 255) / 256; if (blocks > 4096) blocks = 4096;
    f32_to_bf16_kernel<<<blocks, 256, 0, stream>>>(src, dst, n);
  };
  for (int i = 0; i < NLAYERS; ++i) {
    const int p = LBASE + i * LSTRIDE;
    conv((const float*)d_in[p + 2],  wqb[i], D_MODEL * D_MODEL);
    conv((const float*)d_in[p + 4],  wkb[i], D_MODEL * D_MODEL);
    conv((const float*)d_in[p + 6],  wvb[i], D_MODEL * D_MODEL);
    conv((const float*)d_in[p + 10], w1b[i], D_MODEL * D_FF);
    conv((const float*)d_in[p + 12], w2b[i], D_FF * D_MODEL);
  }
  conv(Wout, woutb, D_MODEL * VOCAB);

  embed_kernel<<<ROWS, D_MODEL, 0, stream>>>(x, emb, h, mask);

  const dim3 gD((D_MODEL + BN - 1) / BN, ROWS / BM);   // (4, 32)
  const dim3 gF((D_FF + BN - 1) / BN, ROWS / BM);      // (16, 32)
  const dim3 gAttn(16, 32);

  for (int i = 0; i < NLAYERS; ++i) {
    const int p = LBASE + i * LSTRIDE;
    const float* n1g = (const float*)d_in[p + 0];
    const float* n1b = (const float*)d_in[p + 1];
    const float* bq  = (const float*)d_in[p + 3];
    const float* bk_ = (const float*)d_in[p + 5];
    const float* bv_ = (const float*)d_in[p + 7];
    const float* n2g = (const float*)d_in[p + 8];
    const float* n2b = (const float*)d_in[p + 9];
    const float* b1  = (const float*)d_in[p + 11];
    const float* b2  = (const float*)d_in[p + 13];

    ln_bf16_kernel<<<ROWS / 8, 256, 0, stream>>>(h, n1g, n1b, hln);
    gemm_bf16_kernel<<<gD, 256, 0, stream>>>(hln, wqb[i], bq,  nullptr, qb,
                                             D_MODEL, D_MODEL, 0, 0);
    gemm_bf16_kernel<<<gD, 256, 0, stream>>>(hln, wkb[i], bk_, nullptr, kb,
                                             D_MODEL, D_MODEL, 0, 0);
    gemm_bf16_kernel<<<gD, 256, 0, stream>>>(hln, wvb[i], bv_, nullptr, vb,
                                             D_MODEL, D_MODEL, 0, 0);
    rope_kernel<<<ROWS / 2, 256, 0, stream>>>(qb, kb);
    attn_chunk_kernel<<<gAttn, 1024, 0, stream>>>(kb, vb, mask, kvc, zc);
    attn_prefix_kernel<<<32, 1024, 0, stream>>>(kvc, zc);
    attn_scan_kernel<<<gAttn, 1024, 0, stream>>>(qb, kb, vb, mask, kvc, zc, h);
    ln_bf16_kernel<<<ROWS / 8, 256, 0, stream>>>(h, n2g, n2b, hln);
    gemm_bf16_kernel<<<gF, 256, 0, stream>>>(hln, w1b[i], b1, nullptr, ff1,
                                             D_FF, D_MODEL, 1, 1);
    gemm_bf16_kernel<<<gD, 256, 0, stream>>>(ff1, w2b[i], b2, h, h,
                                             D_MODEL, D_FF, 0, 0);
  }

  ln_bf16_kernel<<<ROWS / 8, 256, 0, stream>>>(h, out_norm_g, out_norm_b, hln);
  const dim3 gOut((VOCAB + BN - 1) / BN, ROWS / BM);   // (1025, 32)
  gemm_bf16_kernel<<<gOut, 256, 0, stream>>>(hln, woutb, bout, nullptr,
                                             (float*)d_out, VOCAB, D_MODEL, 0, 0);
}
